// ListenAttendSpell_38010460570014
// MI455X (gfx1250) — matrix operand order repaired and checked by reading, not compiled, not run on hardware
//
#include <hip/hip_runtime.h>

// ---------------------------------------------------------------------------
// Listen-Attend-Spell forward for gfx1250 (MI455X).
//   1) k_cvt_f16   : fp32 -> f16 weight conversion.
//   2) k_gemm_wmma : C = act(A @ W^T + bias), f16 WMMA, f32 or f16 output.
//                    Vectorized unguarded tile staging (all K % 32 == 0).
//   3) k_lstm_scan : persistent BiLSTM scan; c-state in registers, h in LDS,
//                    per-step z-slice DMA'd into double-buffered LDS by the
//                    Tensor Data Mover (s_wait_tensorcnt sync), Whh streamed
//                    from L2 as WMMA B-fragments.
//   4) k_tokproj   : teacher-forced one-hot token projection (column gather).
//   5) k_decoder   : persistent attention decoder, WMMA for LSTM gates + phi.
// ---------------------------------------------------------------------------

typedef __attribute__((ext_vector_type(16))) _Float16 v16h;
typedef __attribute__((ext_vector_type(8)))  _Float16 v8h;
typedef __attribute__((ext_vector_type(8)))  float    v8f;
typedef __attribute__((ext_vector_type(4)))  unsigned int v4u;
typedef __attribute__((ext_vector_type(8)))  int v8i;
typedef __attribute__((ext_vector_type(4)))  int v4i;

#if defined(__AMDGCN__) && defined(__gfx1250__) && __has_builtin(__builtin_amdgcn_tensor_load_to_lds)
#define USE_TDM 1
#else
#define USE_TDM 0
#endif

__device__ __forceinline__ v16h cat8(v8h lo, v8h hi) {
    v16h r;
#pragma unroll
    for (int i = 0; i < 8; ++i) { r[i] = lo[i]; r[i + 8] = hi[i]; }
    return r;
}

// A-fragment (16x32 f16, M x K): lane L holds row m = L&15; halves are
// K = 16*(L>>4) + e (contiguous 16 halfs).
__device__ __forceinline__ v16h frag_a_lds(const _Float16* base, int stride,
                                           int mbase, int kbase, int lane) {
    int m  = mbase + (lane & 15);
    int hi = lane >> 4;
    const v8h* p0 = (const v8h*)(base + m * stride + kbase + 16 * hi);
    const v8h* p1 = (const v8h*)(base + m * stride + kbase + 16 * hi + 8);
    return cat8(*p0, *p1);
}

// B-fragment (32x16 f16, K x N): lane L holds column n = L&15; halves are
// K = 16*(L>>4) + e (contiguous 16 halfs).
__device__ __forceinline__ v16h frag_b_lds(const _Float16* base, int stride,
                                           int nbase, int kbase, int lane) {
    int n  = nbase + (lane & 15);
    int hi = lane >> 4;
    const v8h* p = (const v8h*)(base + n * stride + kbase + 16 * hi);
    return cat8(p[0], p[1]);
}

// B-fragment from global f16 weights stored [N, K] row-major (L2-resident).
__device__ __forceinline__ v16h frag_b_glb(const _Float16* W, int ldk,
                                           int nbase, int kbase, int lane) {
    const _Float16* p = W + (size_t)(nbase + (lane & 15)) * ldk + kbase + 16 * (lane >> 4);
    const v8h* q = (const v8h*)p;
    return cat8(q[0], q[1]);
}

__device__ __forceinline__ v8f wmma_f16(v16h a, v16h b, v8f c) {
    return __builtin_amdgcn_wmma_f32_16x16x32_f16(false, a, false, b,
                                                  (short)0, c, false, false);
}

__device__ __forceinline__ float sigm(float x) { return 1.f / (1.f + __expf(-x)); }

// ---------------------------------------------------------------------------
// 1) fp32 -> f16 conversion
// ---------------------------------------------------------------------------
__global__ void k_cvt_f16(const float* __restrict__ src, _Float16* __restrict__ dst, int n) {
    int i = blockIdx.x * 256 + threadIdx.x;
    if (i < n) dst[i] = (_Float16)src[i];
}

// ---------------------------------------------------------------------------
// 2) WMMA GEMM: C[M,N] = act(A[M,K] @ W[N,K]^T + bias)
//    Requires K % 32 == 0 (holds: 160/1024/512), M % 64 == 0, N % 128 == 0
//    (N=1024) or grid.x==1 (N=128). Block 256 threads = 8 waves; block tile
//    64x128; wave tile 32x32; K step 32. Output f32 (C) or f16 (C16).
// ---------------------------------------------------------------------------
#define GEMM_LDA 40
#define GEMM_LDB 40
__global__ __launch_bounds__(256) void k_gemm_wmma(
        const float* __restrict__ A, const _Float16* __restrict__ W,
        const float* __restrict__ bias, float* __restrict__ C,
        _Float16* __restrict__ C16, int M, int N, int K, int act) {
    __shared__ _Float16 As[64 * GEMM_LDA];
    __shared__ _Float16 Ws[128 * GEMM_LDB];

    int tid  = threadIdx.x;
    int lane = tid & 31;
    int wid  = tid >> 5;
    int bm = blockIdx.y * 64;
    int bn = blockIdx.x * 128;
    int wm = (wid >> 2) * 32;
    int wn = (wid & 3) * 32;

    // Staging coordinates (vectorized, no tail guards: K % 32 == 0).
    int ar  = tid >> 2, ac = (tid & 3) * 8;   // A: 8 f32 per thread
    int wr  = tid >> 1, wc = (tid & 1) * 16;  // W: 16 f16 per thread

    v8f acc[2][2];
#pragma unroll
    for (int i = 0; i < 2; ++i)
#pragma unroll
        for (int j = 0; j < 2; ++j)
#pragma unroll
            for (int r = 0; r < 8; ++r) acc[i][j][r] = 0.f;

    for (int k0 = 0; k0 < K; k0 += 32) {
        {   // A tile: 64x32 f32 -> f16, one b128 LDS store per thread.
            const float4* ap = (const float4*)(A + (size_t)(bm + ar) * K + k0 + ac);
            float4 a0 = ap[0], a1 = ap[1];
            v8h h;
            h[0] = (_Float16)a0.x; h[1] = (_Float16)a0.y;
            h[2] = (_Float16)a0.z; h[3] = (_Float16)a0.w;
            h[4] = (_Float16)a1.x; h[5] = (_Float16)a1.y;
            h[6] = (_Float16)a1.z; h[7] = (_Float16)a1.w;
            *(v8h*)(As + ar * GEMM_LDA + ac) = h;
        }
        {   // W tile: 128x32 f16, two b128 loads/stores per thread.
            const v8h* wp = (const v8h*)(W + (size_t)(bn + wr) * K + k0 + wc);
            v8h w0 = wp[0], w1 = wp[1];
            *(v8h*)(Ws + wr * GEMM_LDB + wc)     = w0;
            *(v8h*)(Ws + wr * GEMM_LDB + wc + 8) = w1;
        }
        if (k0 + 32 < K) {  // hint next tiles (global_prefetch_b8)
            __builtin_prefetch(A + (size_t)(bm + ar) * K + k0 + 32 + ac, 0, 1);
            __builtin_prefetch(W + (size_t)(bn + wr) * K + k0 + 32 + wc, 0, 1);
        }
        __syncthreads();

        v16h a0 = frag_a_lds(As, GEMM_LDA, wm, 0, lane);
        v16h a1 = frag_a_lds(As, GEMM_LDA, wm + 16, 0, lane);
        v16h b0 = frag_b_lds(Ws, GEMM_LDB, wn, 0, lane);
        v16h b1 = frag_b_lds(Ws, GEMM_LDB, wn + 16, 0, lane);
        acc[0][0] = wmma_f16(a0, b0, acc[0][0]);
        acc[0][1] = wmma_f16(a0, b1, acc[0][1]);
        acc[1][0] = wmma_f16(a1, b0, acc[1][0]);
        acc[1][1] = wmma_f16(a1, b1, acc[1][1]);
        __syncthreads();
    }

    int hi = lane >> 4, ln = lane & 15;
#pragma unroll
    for (int tm = 0; tm < 2; ++tm)
#pragma unroll
        for (int tn = 0; tn < 2; ++tn) {
            int col = bn + wn + tn * 16 + ln;
#pragma unroll
            for (int r = 0; r < 8; ++r) {
                int m = bm + wm + tm * 16 + r + 8 * hi;
                float v = acc[tm][tn][r];
                if (bias) v += bias[col];
                if (act) v = v > 0.f ? v : 0.f;
                if (C16) C16[(size_t)m * N + col] = (_Float16)v;
                else     C[(size_t)m * N + col] = v;
            }
        }
}

// ---------------------------------------------------------------------------
// 3) Persistent BiLSTM scan. grid = 2 blocks (dir 0 = fwd, 1 = bwd),
//    1024 threads = 32 waves. H=256, G=1024, B=64.
//    Wave w: hidden tile ht = w&15 (same cols of all 4 gates -> lane-local
//    gate math), row half mh = w>>4. c-state in VGPRs for all steps; h
//    rebroadcast via LDS f16; per-step z-slice (64x1024 f16 = 128KB) DMA'd
//    into double-buffered LDS by the TDM, overlapped with the WMMA K-loop.
// ---------------------------------------------------------------------------
#define HSTR 272   // 256 + 16 halfs padding (16B-aligned rows)
#define XSLICE (64 * 1024)
__global__ __launch_bounds__(1024) void k_lstm_scan(
        const _Float16* __restrict__ xpF, const _Float16* __restrict__ xpB,
        const _Float16* __restrict__ WhhF, const _Float16* __restrict__ WhhB,
        float* __restrict__ feats, int Tst, float* __restrict__ hc) {
    extern __shared__ char smem[];
    _Float16* hbuf  = (_Float16*)smem;        // [64][HSTR]
    _Float16* xbuf0 = hbuf + 64 * HSTR;       // [2][64][1024] (TDM double buffer)

    const int H = 256, G = 1024;
    int dir = blockIdx.x;
    const _Float16* xp  = dir ? xpB : xpF;
    const _Float16* Whh = dir ? WhhB : WhhF;
    int dirOff = dir * H;

    int tid = threadIdx.x, lane = tid & 31, wid = tid >> 5;
    int ht = wid & 15;
    int mh = wid >> 4;
    int mbase = mh * 32;
    int hi = lane >> 4, ln = lane & 15;

    for (int i = tid; i < 64 * HSTR; i += 1024) hbuf[i] = (_Float16)0.f;

#if USE_TDM
    // 2D TDM descriptor: tile = 1024 x 64 f16 elements, tensor row stride =
    // Tst*1024 elements; dest = LDS slice buffer. Issued by wave 0 only.
    auto issue_tdm = [&](int buf, int tt) {
        unsigned ldsoff = (unsigned)(size_t)(xbuf0 + (size_t)buf * XSLICE);
        unsigned long long ga = (unsigned long long)(size_t)(xp + (size_t)tt * G);
        unsigned long long strideE = (unsigned long long)Tst * G;  // dim0 stride (elems)
        v4u g0;
        g0[0] = 1u;                                   // count=1, user descriptor
        g0[1] = ldsoff;                               // lds_addr
        g0[2] = (unsigned)ga;                         // global_addr[31:0]
        g0[3] = (unsigned)((ga >> 32) & 0x01FFFFFFu) | (2u << 30);  // addr hi | type=2
        v8i g1;
        g1[0] = 0x00010000;                           // data_size = 2 bytes
        g1[1] = (int)(0x0400u << 16);                 // tensor_dim0 = 1024 (lo16)
        g1[2] = (int)(64u << 16);                     // tensor_dim1 = 64 (lo16)
        g1[3] = (int)(1024u << 16);                   // tile_dim0 = 1024
        g1[4] = 64;                                   // tile_dim1 = 64
        g1[5] = (int)(unsigned)(strideE & 0xFFFFFFFFu);       // dim0_stride lo32
        g1[6] = (int)(unsigned)((strideE >> 32) & 0xFFFFu);   // dim0_stride hi16
        g1[7] = 0;
        v4i gz4; gz4[0] = 0; gz4[1] = 0; gz4[2] = 0; gz4[3] = 0;
        v8i gz8;
#pragma unroll
        for (int i = 0; i < 8; ++i) gz8[i] = 0;
        // 6-arg toolchain form: (g0, g1, g2, g3, g4, cpol)
        __builtin_amdgcn_tensor_load_to_lds(g0, g1, gz4, gz4, gz8, 0);
    };
    if (tid < 32) issue_tdm(0, dir ? Tst - 1 : 0);
#endif

    v8f c_st[2];
#pragma unroll
    for (int mt = 0; mt < 2; ++mt)
#pragma unroll
        for (int r = 0; r < 8; ++r) c_st[mt][r] = 0.f;

    for (int t = 0; t < Tst; ++t) {
        int tt = dir ? (Tst - 1 - t) : t;
        int cur = t & 1;

#if USE_TDM
        if (tid < 32) __builtin_amdgcn_s_wait_tensorcnt(0);  // slice[cur] landed
#endif
        __syncthreads();  // h(t-1) visible to all; z-slice ready
#if USE_TDM
        if (t + 1 < Tst && tid < 32)
            issue_tdm(1 - cur, dir ? (Tst - 2 - t) : (t + 1));  // overlap next DMA
        const _Float16* xl = xbuf0 + (size_t)cur * XSLICE;
#endif

        // z tiles: init from x-projection (+bias folded in GEMM).
        v8f acc[2][4];
#pragma unroll
        for (int mt = 0; mt < 2; ++mt)
#pragma unroll
            for (int g = 0; g < 4; ++g) {
                int colb = g * H + ht * 16 + ln;
#pragma unroll
                for (int r = 0; r < 8; ++r) {
                    int m = mbase + mt * 16 + r + 8 * hi;
#if USE_TDM
                    acc[mt][g][r] = (float)xl[m * G + colb];
#else
                    acc[mt][g][r] = (float)xp[((size_t)m * Tst + tt) * G + colb];
#endif
                }
            }

        // z += h @ Whh^T  (K=256, 8 WMMA k-steps; B frags stream from L2).
#pragma unroll
        for (int ks = 0; ks < 8; ++ks) {
            v16h a0 = frag_a_lds(hbuf, HSTR, mbase, ks * 32, lane);
            v16h a1 = frag_a_lds(hbuf, HSTR, mbase + 16, ks * 32, lane);
#pragma unroll
            for (int g = 0; g < 4; ++g) {
                v16h bf = frag_b_glb(Whh, 256, g * H + ht * 16, ks * 32, lane);
                acc[0][g] = wmma_f16(a0, bf, acc[0][g]);
                acc[1][g] = wmma_f16(a1, bf, acc[1][g]);
            }
        }
        __syncthreads();  // everyone done reading old h (and z-slice)

        // Gate nonlinearity + state update (lane-local across the 4 gate tiles).
#pragma unroll
        for (int mt = 0; mt < 2; ++mt) {
#pragma unroll
            for (int r = 0; r < 8; ++r) {
                float iv = sigm(acc[mt][0][r]);
                float fv = sigm(acc[mt][1][r]);
                float gv = tanhf(acc[mt][2][r]);
                float ov = sigm(acc[mt][3][r]);
                float cv = fv * c_st[mt][r] + iv * gv;
                c_st[mt][r] = cv;
                float hv = ov * tanhf(cv);
                int m = mbase + mt * 16 + r + 8 * hi;
                int col = ht * 16 + ln;
                hbuf[m * HSTR + col] = (_Float16)hv;
                feats[((size_t)m * Tst + tt) * 512 + dirOff + col] = hv;
            }
        }
    }
    __syncthreads();

    if (hc) {  // final hT / cT for listener_hc: [B, hF|hB|cF|cB] = [64,1024]
#pragma unroll
        for (int mt = 0; mt < 2; ++mt)
#pragma unroll
            for (int r = 0; r < 8; ++r) {
                int m = mbase + mt * 16 + r + 8 * hi;
                int col = ht * 16 + ln;
                hc[(size_t)m * 1024 + dirOff + col] = (float)hbuf[m * HSTR + col];
                hc[(size_t)m * 1024 + 512 + dirOff + col] = c_st[mt][r];
            }
    }
}

// ---------------------------------------------------------------------------
// 4) Token projection: tokproj[s][b][:] = s_b + s_Wih[:, tok(s,b)]
// ---------------------------------------------------------------------------
__global__ void k_tokproj(const int* __restrict__ gt, const float* __restrict__ sWih,
                          const float* __restrict__ sb, float* __restrict__ tp) {
    size_t i = (size_t)blockIdx.x * 256 + threadIdx.x;
    if (i >= (size_t)150 * 64 * 2048) return;
    int n = (int)(i & 2047);
    size_t sbi = i >> 11;
    int b = (int)(sbi & 63);
    int s = (int)(sbi >> 6);
    int tok = (s == 0) ? 0 : gt[b * 150 + (s - 1)];
    tp[i] = sb[n] + sWih[(size_t)n * 560 + tok];
}

// ---------------------------------------------------------------------------
// 5) Persistent attention decoder. Single block, 1024 threads = 32 waves.
// ---------------------------------------------------------------------------
#define DSTR 528  // 512 + 16 halfs padding
__global__ __launch_bounds__(1024) void k_decoder(
        const float* __restrict__ tokproj, const _Float16* __restrict__ sWih16,
        const _Float16* __restrict__ sWhh16, const _Float16* __restrict__ phiW16,
        const float* __restrict__ phib, const float* __restrict__ comp_feat,
        const float* __restrict__ feats2, const float* __restrict__ fcW,
        const float* __restrict__ fcb, float* __restrict__ out_lps) {
    extern __shared__ char smem2[];
    _Float16* h16  = (_Float16*)smem2;          // [64][DSTR]
    _Float16* cx16 = h16 + 64 * DSTR;           // [64][DSTR]
    float* eatt = (float*)(cx16 + 64 * DSTR);   // [64][256]
    float* qbuf = eatt + 64 * 256;              // [64][132]
    float* lbuf = qbuf + 64 * 132;              // [64][48]
    float* red  = lbuf + 64 * 48;               // [64][2]

    const int SH = 512, TT = 256, V = 48, TD = 150;
    int tid = threadIdx.x, lane = tid & 31, wid = tid >> 5;
    int hi = lane >> 4, ln = lane & 15;
    int htile = wid;  // hidden cols htile*16..+16 of every gate

    for (int i = tid; i < 64 * DSTR; i += 1024) h16[i] = (_Float16)0.f;
    for (int i = tid; i < 64 * 512; i += 1024) {
        int b = i >> 9, d = i & 511;
        cx16[b * DSTR + d] = (_Float16)feats2[((size_t)b * TT) * 512 + d];  // ctx0
    }
    v8f c_st[4];
#pragma unroll
    for (int mt = 0; mt < 4; ++mt)
#pragma unroll
        for (int r = 0; r < 8; ++r) c_st[mt][r] = 0.f;
    __syncthreads();

    for (int s = 0; s < TD; ++s) {
        // ---- z = tokproj[s] + ctx@Wih_ctx^T + h@Whh^T ----
        v8f acc[4][4];
#pragma unroll
        for (int mt = 0; mt < 4; ++mt)
#pragma unroll
            for (int g = 0; g < 4; ++g) {
                int colb = g * SH + htile * 16 + ln;
#pragma unroll
                for (int r = 0; r < 8; ++r) {
                    int m = mt * 16 + r + 8 * hi;
                    acc[mt][g][r] = tokproj[((size_t)s * 64 + m) * 2048 + colb];
                }
            }
        for (int ks = 0; ks < 16; ++ks) {  // ctx part (Wih cols 48..560)
            v16h a[4];
#pragma unroll
            for (int mt = 0; mt < 4; ++mt) a[mt] = frag_a_lds(cx16, DSTR, mt * 16, ks * 32, lane);
#pragma unroll
            for (int g = 0; g < 4; ++g) {
                v16h bf = frag_b_glb(sWih16 + 48, 560, g * SH + htile * 16, ks * 32, lane);
#pragma unroll
                for (int mt = 0; mt < 4; ++mt) acc[mt][g] = wmma_f16(a[mt], bf, acc[mt][g]);
            }
        }
        for (int ks = 0; ks < 16; ++ks) {  // h part
            v16h a[4];
#pragma unroll
            for (int mt = 0; mt < 4; ++mt) a[mt] = frag_a_lds(h16, DSTR, mt * 16, ks * 32, lane);
#pragma unroll
            for (int g = 0; g < 4; ++g) {
                v16h bf = frag_b_glb(sWhh16, 512, g * SH + htile * 16, ks * 32, lane);
#pragma unroll
                for (int mt = 0; mt < 4; ++mt) acc[mt][g] = wmma_f16(a[mt], bf, acc[mt][g]);
            }
        }
        __syncthreads();  // done reading old h / ctx

#pragma unroll
        for (int mt = 0; mt < 4; ++mt)
#pragma unroll
            for (int r = 0; r < 8; ++r) {
                float iv = sigm(acc[mt][0][r]);
                float fv = sigm(acc[mt][1][r]);
                float gv = tanhf(acc[mt][2][r]);
                float ov = sigm(acc[mt][3][r]);
                float cv = fv * c_st[mt][r] + iv * gv;
                c_st[mt][r] = cv;
                float hv = ov * tanhf(cv);
                int m = mt * 16 + r + 8 * hi;
                h16[m * DSTR + htile * 16 + ln] = (_Float16)hv;
            }
        __syncthreads();

        // ---- q = relu(h @ phiW^T + phi_b), one WMMA tile per wave ----
        {
            int mq = (wid >> 3) * 16;
            int nq = (wid & 7) * 16;
            v8f qa;
#pragma unroll
            for (int r = 0; r < 8; ++r) qa[r] = phib[nq + ln];
            for (int ks = 0; ks < 16; ++ks) {
                v16h a  = frag_a_lds(h16, DSTR, mq, ks * 32, lane);
                v16h bf = frag_b_glb(phiW16, 512, nq, ks * 32, lane);
                qa = wmma_f16(a, bf, qa);
            }
#pragma unroll
            for (int r = 0; r < 8; ++r) {
                int m = mq + r + 8 * hi;
                float v = qa[r];
                qbuf[m * 132 + nq + ln] = v > 0.f ? v : 0.f;
            }
        }
        __syncthreads();

        // ---- attention scores, softmax ----
        for (int p = 0; p < 16; ++p) {
            int idx = tid + p * 1024;
            int b = idx >> 8, t = idx & 255;
            const float* cf = &comp_feat[((size_t)b * TT + t) * 128];
            const float* q  = &qbuf[b * 132];
            float sum = 0.f;
            for (int k = 0; k < 128; ++k) sum += q[k] * cf[k];
            eatt[b * 256 + t] = sum;
        }
        __syncthreads();
        if (tid < 64) {
            float m = -3.4e38f;
            for (int t = 0; t < 256; ++t) m = fmaxf(m, eatt[tid * 256 + t]);
            float sum = 0.f;
            for (int t = 0; t < 256; ++t) sum += __expf(eatt[tid * 256 + t] - m);
            red[tid * 2] = m;
            red[tid * 2 + 1] = sum;
        }
        __syncthreads();
        for (int p = 0; p < 16; ++p) {
            int idx = tid + p * 1024;
            int b = idx >> 8, t = idx & 255;
            eatt[b * 256 + t] = __expf(eatt[b * 256 + t] - red[b * 2]) / red[b * 2 + 1];
        }
        __syncthreads();

        // ---- ctx[b,d] = sum_t att[b,t] * feats[b,t,d] ----
        for (int p = 0; p < 32; ++p) {
            int idx = tid + p * 1024;
            int b = idx >> 9, d = idx & 511;
            const float* fr = &feats2[(size_t)b * TT * 512 + d];
            const float* ar = &eatt[b * 256];
            float sum = 0.f;
            for (int t = 0; t < 256; ++t) sum += ar[t] * fr[(size_t)t * 512];
            cx16[b * DSTR + d] = (_Float16)sum;
        }
        __syncthreads();

        // ---- logits + log_softmax ----
        for (int p = 0; p < 3; ++p) {
            int idx = tid + p * 1024;
            if (idx < 64 * V) {
                int b = idx / V, v = idx % V;
                const float* wr = &fcW[(size_t)v * 1024];
                float sum = fcb[v];
                for (int k = 0; k < 512; ++k) sum += (float)h16[b * DSTR + k] * wr[k];
                for (int k = 0; k < 512; ++k) sum += (float)cx16[b * DSTR + k] * wr[512 + k];
                lbuf[b * V + v] = sum;
            }
        }
        __syncthreads();
        if (tid < 64) {
            float m = -3.4e38f;
            for (int v = 0; v < V; ++v) m = fmaxf(m, lbuf[tid * V + v]);
            float sum = 0.f;
            for (int v = 0; v < V; ++v) sum += __expf(lbuf[tid * V + v] - m);
            red[tid * 2] = m;
            red[tid * 2 + 1] = logf(sum);
        }
        __syncthreads();
        for (int p = 0; p < 3; ++p) {
            int idx = tid + p * 1024;
            if (idx < 64 * V) {
                int b = idx / V, v = idx % V;
                out_lps[((size_t)b * TD + s) * V + v] = lbuf[b * V + v] - red[b * 2] - red[b * 2 + 1];
            }
        }
        __syncthreads();
    }
}

// ---------------------------------------------------------------------------
// Host orchestration
// ---------------------------------------------------------------------------
extern "C" void kernel_launch(void* const* d_in, const int* in_sizes, int n_in,
                              void* d_out, int out_size, void* d_ws, size_t ws_size,
                              hipStream_t stream) {
    (void)in_sizes; (void)n_in; (void)out_size; (void)ws_size;

    const float* inputs = (const float*)d_in[0];
    const int*   gt     = (const int*)d_in[1];
    const float* lWih[3][2] = {{(const float*)d_in[2],  (const float*)d_in[5]},
                               {(const float*)d_in[8],  (const float*)d_in[11]},
                               {(const float*)d_in[14], (const float*)d_in[17]}};
    const float* lWhh[3][2] = {{(const float*)d_in[3],  (const float*)d_in[6]},
                               {(const float*)d_in[9],  (const float*)d_in[12]},
                               {(const float*)d_in[15], (const float*)d_in[18]}};
    const float* lB[3][2]   = {{(const float*)d_in[4],  (const float*)d_in[7]},
                               {(const float*)d_in[10], (const float*)d_in[13]},
                               {(const float*)d_in[16], (const float*)d_in[19]}};
    const float* sWih = (const float*)d_in[20];
    const float* sWhh = (const float*)d_in[21];
    const float* sB   = (const float*)d_in[22];
    const float* phiW = (const float*)d_in[23];
    const float* phiB = (const float*)d_in[24];
    const float* psiW = (const float*)d_in[25];
    const float* psiB = (const float*)d_in[26];
    const float* fcW  = (const float*)d_in[27];
    const float* fcB  = (const float*)d_in[28];

    char* ws = (char*)d_ws;
    size_t off = 0;
    auto alloc = [&](size_t bytes) -> char* {
        char* p = ws + off;
        off += (bytes + 255) & ~(size_t)255;
        return p;
    };

    // f16 weight copies
    const int wihN[3] = {1024 * 160, 1024 * 1024, 1024 * 1024};
    _Float16* wih16[3][2];
    _Float16* whh16[3][2];
    for (int l = 0; l < 3; ++l)
        for (int d = 0; d < 2; ++d) {
            wih16[l][d] = (_Float16*)alloc((size_t)wihN[l] * 2);
            whh16[l][d] = (_Float16*)alloc((size_t)1024 * 256 * 2);
        }
    _Float16* sWih16 = (_Float16*)alloc((size_t)2048 * 560 * 2);
    _Float16* sWhh16 = (_Float16*)alloc((size_t)2048 * 512 * 2);
    _Float16* phiW16 = (_Float16*)alloc((size_t)128 * 512 * 2);
    _Float16* psiW16 = (_Float16*)alloc((size_t)128 * 512 * 2);

    // activation buffers (x-projections f16 -> TDM-friendly + half BW)
    _Float16* xpA = (_Float16*)alloc((size_t)65536 * 1024 * 2);
    _Float16* xpB = (_Float16*)alloc((size_t)65536 * 1024 * 2);
    float* feats0 = (float*)alloc((size_t)64 * 1024 * 512 * 4);
    float* feats1 = (float*)alloc((size_t)64 * 512 * 512 * 4);
    float* feats2 = (float*)alloc((size_t)64 * 256 * 512 * 4);
    float* comp   = (float*)alloc((size_t)64 * 256 * 128 * 4);
    float* tokp   = (float*)alloc((size_t)150 * 64 * 2048 * 4);

    auto cvt = [&](const float* s, _Float16* d, size_t n) {
        k_cvt_f16<<<dim3((unsigned)((n + 255) / 256)), dim3(256), 0, stream>>>(s, d, (int)n);
    };
    for (int l = 0; l < 3; ++l)
        for (int d = 0; d < 2; ++d) {
            cvt(lWih[l][d], wih16[l][d], (size_t)wihN[l]);
            cvt(lWhh[l][d], whh16[l][d], (size_t)1024 * 256);
        }
    cvt(sWih, sWih16, (size_t)2048 * 560);
    cvt(sWhh, sWhh16, (size_t)2048 * 512);
    cvt(phiW, phiW16, (size_t)128 * 512);
    cvt(psiW, psiW16, (size_t)128 * 512);

    // Encoder: pyramid BiLSTM ((B,T,d)->(B,T/2,2d) is a pure view).
    const float* layerA[3] = {inputs, feats0, feats1};
    float* layerF[3] = {feats0, feats1, feats2};
    const int layerT[3] = {1024, 512, 256};
    const int layerK[3] = {160, 1024, 1024};
    float* hcOut = (float*)d_out + (size_t)64 * 150 * 48;

    size_t scanLds = ((size_t)64 * HSTR + 2 * (size_t)XSLICE) * 2;  // hbuf + TDM dbl buf

    for (int l = 0; l < 3; ++l) {
        int M = 64 * layerT[l];
        dim3 grid(1024 / 128, M / 64);
        k_gemm_wmma<<<grid, dim3(256), 0, stream>>>(layerA[l], wih16[l][0], lB[l][0],
                                                    nullptr, xpA, M, 1024, layerK[l], 0);
        k_gemm_wmma<<<grid, dim3(256), 0, stream>>>(layerA[l], wih16[l][1], lB[l][1],
                                                    nullptr, xpB, M, 1024, layerK[l], 0);
        k_lstm_scan<<<dim3(2), dim3(1024), scanLds, stream>>>(
            xpA, xpB, whh16[l][0], whh16[l][1], layerF[l], layerT[l],
            (l == 2) ? hcOut : nullptr);
    }

    // comp_feat = relu(feats2 @ psiW^T + psi_b)
    k_gemm_wmma<<<dim3(1, 16384 / 64), dim3(256), 0, stream>>>(
        feats2, psiW16, psiB, comp, nullptr, 16384, 128, 512, 1);

    // Teacher-forced token projection.
    {
        size_t tot = (size_t)150 * 64 * 2048;
        k_tokproj<<<dim3((unsigned)((tot + 255) / 256)), dim3(256), 0, stream>>>(
            gt, sWih, sB, tokp);
    }

    // Attention decoder (persistent single workgroup).
    size_t dls = (size_t)64 * DSTR * 2 * 2
               + ((size_t)64 * 256 + 64 * 132 + 64 * 48 + 128) * 4;
    k_decoder<<<dim3(1), dim3(1024), dls, stream>>>(
        tokp, sWih16, sWhh16, phiW16, phiB, comp, feats2, fcW, fcB, (float*)d_out);
}
